// DeepseekMoE_21921513078943
// MI455X (gfx1250) — matrix operand order repaired and checked by reading, not compiled, not run on hardware
//
#include <hip/hip_runtime.h>
#include <hip/hip_bf16.h>

// Problem dims (fixed by the reference)
#define DD    1024            // hidden
#define II    512             // intermediate
#define EE    16              // experts
#define NTOK  4096            // B*S

typedef __bf16 bf16t;
typedef __attribute__((ext_vector_type(16))) __bf16 v16bf;
typedef __attribute__((ext_vector_type(8)))  float  v8f;

__device__ inline unsigned pack_bf16(float a, float b) {
    union { __bf16 h[2]; unsigned u; } p;
    p.h[0] = (bf16t)a; p.h[1] = (bf16t)b;
    return p.u;
}

__device__ inline float fast_silu(float g) {
    // g * sigmoid(g) via v_rcp_f32 (avoid IEEE divide expansion)
    return g * __builtin_amdgcn_rcpf(1.f + __expf(-g));
}

// ---- WMMA fragment loaders (ISA 7.12.2 bf16 layouts) ----------------------
// A 16x32 (M x K), LDS row-major [16][32] (caller passes m-half base).
__device__ inline v16bf load_a_frag(const bf16t* ldsA, int lane) {
    const int half = lane >> 4, row = lane & 15;
    const bf16t* p = ldsA + row * 32 + half * 8;
    union { v16bf v; uint4 q[2]; } u;
    u.q[0] = *(const uint4*)(p);
    u.q[1] = *(const uint4*)(p + 16);
    return u.v;
}
// B 32x16 (K x N), LDS stored transposed/N-major [64][32] so per-lane K run is contiguous.
__device__ inline v16bf load_b_frag(const bf16t* ldsBt, int nbase, int lane) {
    const int half = lane >> 4, col = lane & 15;
    const bf16t* p = ldsBt + (nbase + col) * 32 + half * 8;
    union { v16bf v; uint4 q[2]; } u;
    u.q[0] = *(const uint4*)(p);
    u.q[1] = *(const uint4*)(p + 16);
    return u.v;
}

#define WMMA_BF16(A, B, C) \
    __builtin_amdgcn_wmma_f32_16x16x32_bf16(false, (A), false, (B), (short)0, (C), false, false)

// ---- init -----------------------------------------------------------------
__global__ void moe_zero_counts(int* __restrict__ counts) {
    if (threadIdx.x < EE) counts[threadIdx.x] = 0;
}

// ---- router: sigmoid logits -> top2 -> normalized weights + expert lists --
__global__ __launch_bounds__(256)
void moe_router(const float* __restrict__ x, const float* __restrict__ gw,
                const float* __restrict__ gb, int* __restrict__ counts,
                int* __restrict__ lists, float* __restrict__ wts) {
    __shared__ float part[256][EE];
    const int t = blockIdx.x, tid = threadIdx.x;
    float acc[EE];
#pragma unroll
    for (int e = 0; e < EE; ++e) acc[e] = 0.f;
    for (int d = tid; d < DD; d += 256) {
        const float xv = x[(size_t)t * DD + d];
        const float* grow = gw + (size_t)d * EE;
#pragma unroll
        for (int e = 0; e < EE; e += 4) {
            float4 g4 = *(const float4*)(grow + e);
            acc[e + 0] += xv * g4.x; acc[e + 1] += xv * g4.y;
            acc[e + 2] += xv * g4.z; acc[e + 3] += xv * g4.w;
        }
    }
#pragma unroll
    for (int e = 0; e < EE; ++e) part[tid][e] = acc[e];
    __syncthreads();
    if (tid < EE) {
        float s = 0.f;
        for (int i = 0; i < 256; ++i) s += part[i][tid];
        part[0][tid] = __builtin_amdgcn_rcpf(1.f + __expf(-(s + gb[tid]))); // sigmoid
    }
    __syncthreads();
    if (tid == 0) {
        float v[EE];
#pragma unroll
        for (int e = 0; e < EE; ++e) v[e] = part[0][e];
        int i1 = 0;
        for (int e = 1; e < EE; ++e) if (v[e] > v[i1]) i1 = e;
        int i2 = (i1 == 0) ? 1 : 0;
        for (int e = 0; e < EE; ++e) if (e != i1 && v[e] > v[i2]) i2 = e;
        const float inv = __builtin_amdgcn_rcpf(v[i1] + v[i2]);
        wts[t * 2 + 0] = v[i1] * inv;
        wts[t * 2 + 1] = v[i2] * inv;
        int s1 = atomicAdd(&counts[i1], 1); lists[i1 * NTOK + s1] = t * 2 + 0;
        int s2 = atomicAdd(&counts[i2], 1); lists[i2 * NTOK + s2] = t * 2 + 1;
    }
}

// ---- stage 1: H = silu(X@Wg) * (X@Wu)   (32M x 64N tile, K-loop 32) -------
__global__ __launch_bounds__(128)
void moe_stage1(const float* __restrict__ x,
                const float* __restrict__ Wg, const float* __restrict__ Wu,
                const int* __restrict__ counts, const int* __restrict__ lists,
                bf16t* __restrict__ Hout, int routed) {
    __shared__ __align__(16) bf16t ldsA[32 * 32];
    __shared__ __align__(16) bf16t ldsBg[64 * 32];
    __shared__ __align__(16) bf16t ldsBu[64 * 32];
    __shared__ int srow[32];   // source token row in x (-1 = padding)
    __shared__ int said[32];   // destination row in Hout (-1 = padding)

    const int e = blockIdx.z;
    int cnt = NTOK;
    const float* wg = Wg; const float* wu = Wu;
    if (routed) {
        cnt = counts[e];
        if ((int)blockIdx.x * 32 >= cnt) return;
        const size_t off = (size_t)e * DD * II;
        wg += off; wu += off;
    }
    const int mbase = blockIdx.x * 32;
    const int n0 = blockIdx.y * 64;
    const int tid = threadIdx.x, lane = tid & 31, wv = tid >> 5;

    if (tid < 32) {
        const int m = mbase + tid;
        if (routed) {
            if (m < cnt) { const int aid = lists[e * NTOK + m]; srow[tid] = aid >> 1; said[tid] = aid; }
            else         { srow[tid] = -1; said[tid] = -1; }
        } else { srow[tid] = m; said[tid] = m; }
    }
    __syncthreads();

    // Clamp padded rows to row 0 of the tile (always valid: mbase < cnt).
    // Garbage rows only affect D rows we never store.
    const int tok0 = srow[0];
    int tokA[2];
#pragma unroll
    for (int j = 0; j < 2; ++j) {
        const int t_ = srow[(tid + j * 128) >> 3];
        tokA[j] = (t_ < 0) ? tok0 : t_;
    }

    v8f cg0 = {}, cg1 = {}, cu0 = {}, cu1 = {};
    for (int k0 = 0; k0 < DD; k0 += 32) {
        // A tile 32x32 (gathered, fp32 -> packed bf16, ds_store_b64)
#pragma unroll
        for (int j = 0; j < 2; ++j) {
            const int lin = tid + j * 128;          // 0..255
            const int ar = lin >> 3, ac = (lin & 7) * 4;
            float4 f = *(const float4*)(x + (size_t)tokA[j] * DD + k0 + ac);
            union { unsigned u[2]; uint2 v; } pk;
            pk.u[0] = pack_bf16(f.x, f.y); pk.u[1] = pack_bf16(f.z, f.w);
            *(uint2*)(ldsA + ar * 32 + ac) = pk.v;
        }
        // B tiles 32K x 64N -> transposed LDS [64][32]; pack K-pairs (ds_store_b32)
#pragma unroll
        for (int j = 0; j < 2; ++j) {
            const int lin = tid + j * 128;          // 0..255
            const int kp = lin >> 4;                // 0..15 -> k rows {2kp, 2kp+1}
            const int cs = (lin & 15) * 4;          // 0..60
            const float* pg = wg + (size_t)(k0 + 2 * kp) * II + n0 + cs;
            const float* pu = wu + (size_t)(k0 + 2 * kp) * II + n0 + cs;
            float4 g0 = *(const float4*)(pg), g1 = *(const float4*)(pg + II);
            float4 u0 = *(const float4*)(pu), u1 = *(const float4*)(pu + II);
            if (k0 + 32 < DD) {                      // prefetch next K tile
                __builtin_prefetch(pg + 32 * II, 0, 0);
                __builtin_prefetch(pu + 32 * II, 0, 0);
            }
            *(unsigned*)(ldsBg + (cs + 0) * 32 + 2 * kp) = pack_bf16(g0.x, g1.x);
            *(unsigned*)(ldsBg + (cs + 1) * 32 + 2 * kp) = pack_bf16(g0.y, g1.y);
            *(unsigned*)(ldsBg + (cs + 2) * 32 + 2 * kp) = pack_bf16(g0.z, g1.z);
            *(unsigned*)(ldsBg + (cs + 3) * 32 + 2 * kp) = pack_bf16(g0.w, g1.w);
            *(unsigned*)(ldsBu + (cs + 0) * 32 + 2 * kp) = pack_bf16(u0.x, u1.x);
            *(unsigned*)(ldsBu + (cs + 1) * 32 + 2 * kp) = pack_bf16(u0.y, u1.y);
            *(unsigned*)(ldsBu + (cs + 2) * 32 + 2 * kp) = pack_bf16(u0.z, u1.z);
            *(unsigned*)(ldsBu + (cs + 3) * 32 + 2 * kp) = pack_bf16(u0.w, u1.w);
        }
        __syncthreads();
        const v16bf a0 = load_a_frag(ldsA, lane);
        const v16bf a1 = load_a_frag(ldsA + 16 * 32, lane);
        const v16bf bg = load_b_frag(ldsBg, wv * 16, lane);
        const v16bf bu = load_b_frag(ldsBu, wv * 16, lane);
        cg0 = WMMA_BF16(a0, bg, cg0);
        cg1 = WMMA_BF16(a1, bg, cg1);
        cu0 = WMMA_BF16(a0, bu, cu0);
        cu1 = WMMA_BF16(a1, bu, cu1);
        __syncthreads();
    }
    // epilogue: h = silu(g)*u stored as bf16
    const int nl = lane & 15, mo = (lane >> 4) * 8;
    const int gn = n0 + wv * 16 + nl;
#pragma unroll
    for (int r = 0; r < 8; ++r) {
        const int m = r + mo;
        int rid = said[m];
        if (rid >= 0)
            Hout[(size_t)rid * II + gn] = (bf16t)(fast_silu(cg0[r]) * cu0[r]);
        rid = said[m + 16];
        if (rid >= 0)
            Hout[(size_t)rid * II + gn] = (bf16t)(fast_silu(cg1[r]) * cu1[r]);
    }
}

// ---- stage 2: out (+)= H @ Wd  (shared: write x+·, routed: atomic w*·) ----
// A tile (bf16 H rows) fetched with ASYNC global->LDS DMA (no VGPR roundtrip).
__global__ __launch_bounds__(128)
void moe_stage2(const bf16t* __restrict__ H, const float* __restrict__ Wd,
                const float* __restrict__ x,
                const int* __restrict__ counts, const int* __restrict__ lists,
                const float* __restrict__ wts,
                float* __restrict__ out, int routed) {
    __shared__ __align__(16) bf16t ldsA[32 * 32];
    __shared__ __align__(16) bf16t ldsB[64 * 32];
    __shared__ int said[32];

    const int e = blockIdx.z;
    int cnt = NTOK;
    const float* wd = Wd;
    if (routed) {
        cnt = counts[e];
        if ((int)blockIdx.x * 32 >= cnt) return;
        wd += (size_t)e * II * DD;
    }
    const int mbase = blockIdx.x * 32;
    const int n0 = blockIdx.y * 64;
    const int tid = threadIdx.x, lane = tid & 31, wv = tid >> 5;

    if (tid < 32) {
        const int m = mbase + tid;
        said[tid] = routed ? ((m < cnt) ? lists[e * NTOK + m] : -1) : m;
    }
    __syncthreads();

    // A copy assignment: row = tid>>2 (0..31), seg = (tid&3)*8; clamp padding to row 0.
    const int arow = tid >> 2, aseg = (tid & 3) * 8;
    int hrow = said[arow];
    hrow = (hrow < 0) ? said[0] : hrow;     // garbage rows never stored
    const bf16t* hsrc = H + (size_t)hrow * II + aseg;
    const unsigned ldsa_off = (unsigned)(uintptr_t)(ldsA + arow * 32 + aseg);

    v8f c0 = {}, c1 = {};
    for (int k0 = 0; k0 < II; k0 += 32) {
        // A tile 32x32 bf16: async DMA global -> LDS (b128 per lane, ASYNCcnt)
        asm volatile("global_load_async_to_lds_b128 %0, %1, off"
                     :: "v"(ldsa_off), "v"(hsrc + k0)
                     : "memory");
#pragma unroll
        for (int j = 0; j < 2; ++j) {       // B tile 32K x 64N -> transposed, packed
            const int lin = tid + j * 128;
            const int kp = lin >> 4, cs = (lin & 15) * 4;
            const float* pb = wd + (size_t)(k0 + 2 * kp) * DD + n0 + cs;
            float4 b0 = *(const float4*)(pb), b1 = *(const float4*)(pb + DD);
            if (k0 + 32 < II)               // prefetch next K tile
                __builtin_prefetch(pb + 32 * DD, 0, 0);
            *(unsigned*)(ldsB + (cs + 0) * 32 + 2 * kp) = pack_bf16(b0.x, b1.x);
            *(unsigned*)(ldsB + (cs + 1) * 32 + 2 * kp) = pack_bf16(b0.y, b1.y);
            *(unsigned*)(ldsB + (cs + 2) * 32 + 2 * kp) = pack_bf16(b0.z, b1.z);
            *(unsigned*)(ldsB + (cs + 3) * 32 + 2 * kp) = pack_bf16(b0.w, b1.w);
        }
        asm volatile("s_wait_asynccnt 0" ::: "memory");
        __syncthreads();
        const v16bf a0 = load_a_frag(ldsA, lane);
        const v16bf a1 = load_a_frag(ldsA + 16 * 32, lane);
        const v16bf b  = load_b_frag(ldsB, wv * 16, lane);
        c0 = WMMA_BF16(a0, b, c0);
        c1 = WMMA_BF16(a1, b, c1);
        __syncthreads();
    }

    const int nl = lane & 15, mo = (lane >> 4) * 8;
    const int gn = n0 + wv * 16 + nl;
#pragma unroll
    for (int r = 0; r < 8; ++r) {
#pragma unroll
        for (int h = 0; h < 2; ++h) {
            const int m = r + mo + h * 16;
            const int rid = said[m];
            const float cv = h ? c1[r] : c0[r];
            if (rid >= 0) {
                if (routed) {
                    const int tok = rid >> 1;
                    atomicAdd(out + (size_t)tok * DD + gn, wts[rid] * cv);
                } else {
                    out[(size_t)rid * DD + gn] = x[(size_t)rid * DD + gn] + cv;
                }
            }
        }
    }
}

extern "C" void kernel_launch(void* const* d_in, const int* in_sizes, int n_in,
                              void* d_out, int out_size, void* d_ws, size_t ws_size,
                              hipStream_t stream) {
    const float* x   = (const float*)d_in[0];
    const float* gw  = (const float*)d_in[1];
    const float* gb  = (const float*)d_in[2];
    const float* swg = (const float*)d_in[3];
    const float* swu = (const float*)d_in[4];
    const float* swd = (const float*)d_in[5];
    const float* ewg = (const float*)d_in[6];
    const float* ewu = (const float*)d_in[7];
    const float* ewd = (const float*)d_in[8];
    float* out = (float*)d_out;

    // workspace layout (~12.9 MB)
    char* ws = (char*)d_ws;
    int*   counts = (int*)ws;                                   // 64 B (pad 256)
    int*   lists  = (int*)(ws + 256);                           // 16*4096*4 = 256 KiB
    float* wts    = (float*)(ws + 256 + 262144);                // 8192*4 = 32 KiB
    bf16t* Hs     = (bf16t*)(ws + 256 + 262144 + 32768);        // 4096*512*2 = 4 MiB
    bf16t* Hr     = (bf16t*)(ws + 256 + 262144 + 32768 + (size_t)NTOK * II * 2); // 8 MiB

    moe_zero_counts<<<1, 32, 0, stream>>>(counts);
    moe_router<<<NTOK, 256, 0, stream>>>(x, gw, gb, counts, lists, wts);
    // shared expert up/gate, then routed experts up/gate (independent)
    moe_stage1<<<dim3(NTOK / 32, II / 64, 1),  128, 0, stream>>>(x, swg, swu, counts, lists, Hs, 0);
    moe_stage1<<<dim3(NTOK / 32, II / 64, EE), 128, 0, stream>>>(x, ewg, ewu, counts, lists, Hr, 1);
    // shared down-proj writes out = x + shared MLP (residual base), then routed scatter-adds
    moe_stage2<<<dim3(NTOK / 32, DD / 64, 1),  128, 0, stream>>>(Hs, swd, x, counts, lists, wts, out, 0);
    moe_stage2<<<dim3(NTOK / 32, DD / 64, EE), 128, 0, stream>>>(Hr, ewd, x, counts, lists, wts, out, 1);
}
